// MultiScaleAttention_85504208929162
// MI455X (gfx1250) — matrix operand order repaired and checked by reading, not compiled, not run on hardware
//
#include <hip/hip_runtime.h>

// ---------------- types ----------------
typedef unsigned short u16;
typedef unsigned int   u32;
typedef unsigned long long u64;
typedef __attribute__((ext_vector_type(16))) __bf16 v16bf;
typedef __attribute__((ext_vector_type(8)))  float  v8f;
typedef __attribute__((ext_vector_type(4)))  unsigned int u32x4;
typedef __attribute__((ext_vector_type(8)))  int i32x8;
typedef __attribute__((ext_vector_type(4)))  int i32x4;

#if defined(__AMDGCN__) && __has_builtin(__builtin_amdgcn_tensor_load_to_lds) && __has_builtin(__builtin_amdgcn_s_wait_tensorcnt)
#define USE_TDM 1
#else
#define USE_TDM 0
#endif

__device__ __forceinline__ u16 f2bf(float f) {
    u32 u = __float_as_uint(f);
    u32 r = u + 0x7FFFu + ((u >> 16) & 1u);   // round-to-nearest-even
    return (u16)(r >> 16);
}
__device__ __forceinline__ float bf2f(u16 h) {
    return __uint_as_float(((u32)h) << 16);
}

// ---------------- problem constants ----------------
#define BB   16
#define HH   56
#define WW   56
#define NN   (HH*WW)          // 3136
#define CC   768
#define MTOT (BB*NN)          // 50176
#define QKVN (3*CC)           // 2304
#define LCC  256
#define HDIM 64

// ---------------- cast f32 -> bf16 ----------------
__global__ void cast_f32_bf16(const float* __restrict__ in, u16* __restrict__ out, long n) {
    long i = (long)blockIdx.x * blockDim.x + threadIdx.x;
    if (i < n) out[i] = f2bf(in[i]);
}

// ---------------- WMMA bf16 GEMM: C[M,N] = A[M,K] * B[N,K]^T (+bias) ----------------
// BM=128, BN=256, BK=32; 256 threads = 8 waves in 2(m) x 4(n); each wave: 4x4 16x16 tiles.
#define BM 128
#define BN 256
#define BK 32

#if USE_TDM
// 2D TDM load: tile (BK elems wide x rows) from row-major src (leading dim ld elems,
// 2 bytes/elem) into contiguous LDS at lds_byte. Issued by one wave; EXEC ignored.
__device__ __forceinline__ void tdm_load_tile(u32 lds_byte, const u16* gsrc,
                                              int rows, int ld_elems)
{
    u64 ga = (u64)(size_t)gsrc;
    u32x4 g0;
    g0.x = 1u;                                 // count=1 (valid user descriptor)
    g0.y = lds_byte;                           // lds_addr [63:32]
    g0.z = (u32)ga;                            // global_addr [95:64]
    g0.w = (u32)(ga >> 32) | (2u << 30);       // global_addr [120:96] | type=2
    i32x8 g1;
    g1[0] = (int)(1u << 16);                   // workgroup_mask=0, data_size=1 (2B)
    g1[1] = (int)(32u << 16);                  // tensor_dim0 = 32 (bits 79:48)
    g1[2] = (int)(((u32)rows) << 16);          // tensor_dim1 lo (bits 111:80)
    g1[3] = (int)((((u32)rows) >> 16) | (32u << 16)); // tensor_dim1 hi | tile_dim0=32
    g1[4] = (int)(u32)rows;                    // tile_dim1 (bits 143:128), tile_dim2=0
    g1[5] = (int)(u32)ld_elems;                // tensor_dim0_stride lo (bits 191:160)
    g1[6] = 0;                                 // stride hi, tensor_dim1_stride lo
    g1[7] = 0;
    i32x4 gz; gz[0] = 0; gz[1] = 0; gz[2] = 0; gz[3] = 0;
    i32x8 gz8;
    gz8[0] = 0; gz8[1] = 0; gz8[2] = 0; gz8[3] = 0;
    gz8[4] = 0; gz8[5] = 0; gz8[6] = 0; gz8[7] = 0;
    // 6-arg form (clang-23 / therock-10.0 headers): extra int32x8 group before cpol.
    __builtin_amdgcn_tensor_load_to_lds(g0, g1, gz, gz, gz8, 0);
}
#endif

union FragU { v16bf v; u32 u[8]; };

__device__ __forceinline__ void compute_tiles(const u16* __restrict__ la,
                                              const u16* __restrict__ lb,
                                              v8f (&acc)[4][4],
                                              int wave_m, int wave_n,
                                              int mrow, int half)
{
    FragU fa[4], fb[4];
    // A fragment: lane holds row m = (lane&15); K pairs interleaved per ISA layout.
    #pragma unroll
    for (int mt = 0; mt < 4; ++mt) {
        int m = wave_m * 64 + mt * 16 + mrow;
        #pragma unroll
        for (int p = 0; p < 8; ++p) {
            int kk = ((p < 4) ? (2 * p) : (16 + 2 * (p - 4))) + 8 * half;
            fa[mt].u[p] = *(const u32*)&la[m * BK + kk];
        }
    }
    // B fragment: lane holds col n = (lane&15); K pairs interleaved per ISA layout (same as A).
    const int kb = half * 8;
    #pragma unroll
    for (int nt = 0; nt < 4; ++nt) {
        int n = wave_n * 64 + nt * 16 + mrow;
        #pragma unroll
        for (int p = 0; p < 8; ++p)
            fb[nt].u[p] = *(const u32*)&lb[n * BK + kb + ((p < 4) ? (2 * p) : (16 + 2 * (p - 4)))];
    }
    #pragma unroll
    for (int mt = 0; mt < 4; ++mt)
        #pragma unroll
        for (int nt = 0; nt < 4; ++nt)
            acc[mt][nt] = __builtin_amdgcn_wmma_f32_16x16x32_bf16(
                false, fa[mt].v, false, fb[nt].v,
                (short)0, acc[mt][nt], false, false);
}

template<bool ADD_BIAS, bool OUT_BF16>
__global__ __launch_bounds__(256) void wmma_gemm_kernel(
    const u16* __restrict__ A, const u16* __restrict__ Bw,
    const float* __restrict__ bias, void* __restrict__ Cout,
    int M, int N, int K)
{
    __shared__ u16 lsA[2][BM * BK];
    __shared__ u16 lsB[2][BN * BK];

    const int tid    = threadIdx.x;
    const int lane   = tid & 31;
    const int wave   = tid >> 5;
    const int wave_m = wave & 1;   // 0..1
    const int wave_n = wave >> 1;  // 0..3
    const int m0 = blockIdx.x * BM;
    const int n0 = blockIdx.y * BN;
    const int mrow = lane & 15;
    const int half = lane >> 4;
    const int steps = K / BK;

    v8f acc[4][4] = {};

#if USE_TDM
    // prologue: TDM loads of buffer 0 (wave 0 -> A tile, wave 1 -> B tile)
    if (wave == 0) tdm_load_tile((u32)(size_t)&lsA[0][0], A  + (size_t)m0 * K, BM, K);
    if (wave == 1) tdm_load_tile((u32)(size_t)&lsB[0][0], Bw + (size_t)n0 * K, BN, K);

    for (int s = 0; s < steps; ++s) {
        const int cur = s & 1;
        __builtin_amdgcn_s_wait_tensorcnt(0);   // issuer waits its tile; others NOP
        __syncthreads();                        // publish buffer `cur` to all waves
        if (s + 1 < steps) {                    // prefetch next k-slab into alt buffer
            const int k1 = (s + 1) * BK;
            if (wave == 0) tdm_load_tile((u32)(size_t)&lsA[cur ^ 1][0], A  + (size_t)m0 * K + k1, BM, K);
            if (wave == 1) tdm_load_tile((u32)(size_t)&lsB[cur ^ 1][0], Bw + (size_t)n0 * K + k1, BN, K);
        }
        compute_tiles(lsA[cur], lsB[cur], acc, wave_m, wave_n, mrow, half);
    }
#else
    for (int s = 0; s < steps; ++s) {
        const int k0 = s * BK;
        for (int i = tid; i < BM * 4; i += 256) {
            int r = i >> 2, seg = (i & 3) * 8;
            *(uint4*)&lsA[0][r * BK + seg] = *(const uint4*)&A[(size_t)(m0 + r) * K + k0 + seg];
        }
        for (int i = tid; i < BN * 4; i += 256) {
            int r = i >> 2, seg = (i & 3) * 8;
            *(uint4*)&lsB[0][r * BK + seg] = *(const uint4*)&Bw[(size_t)(n0 + r) * K + k0 + seg];
        }
        __syncthreads();
        compute_tiles(lsA[0], lsB[0], acc, wave_m, wave_n, mrow, half);
        __syncthreads();
    }
#endif

    // D layout: VGPR e -> M = e + (lane>>4)*8 ; N = lane&15
    #pragma unroll
    for (int mt = 0; mt < 4; ++mt)
        #pragma unroll
        for (int nt = 0; nt < 4; ++nt)
            #pragma unroll
            for (int e = 0; e < 8; ++e) {
                int gm = m0 + wave_m * 64 + mt * 16 + half * 8 + e;
                int gn = n0 + wave_n * 64 + nt * 16 + mrow;
                float val = acc[mt][nt][e];
                if (ADD_BIAS) val += bias[gn];
                if (OUT_BF16) ((u16*)Cout)[(size_t)gm * N + gn] = f2bf(val);
                else          ((float*)Cout)[(size_t)gm * N + gn] = val;
            }
}

// ---------------- scale-1 grouped conv 3x3 (s1,p1,d1) + maxpool2 ----------------
// in: qkv bf16 (B,N,2304), channels [256..512); out g1 bf16 (B,3,256,28,28)
__global__ void conv1_pool_kernel(const u16* __restrict__ qkv,
                                  const float* __restrict__ w,
                                  const float* __restrict__ bias,
                                  u16* __restrict__ out)
{
    long idx = (long)blockIdx.x * blockDim.x + threadIdx.x;
    const long total = (long)BB * 3 * LCC * 28 * 28;
    if (idx >= total) return;
    int wp = idx % 28; long t = idx / 28;
    int hp = t % 28;   t /= 28;
    int oc = t % LCC;  t /= LCC;
    int j  = t % 3;    int b = t / 3;
    int g  = oc >> 2;
    const float* wk = w + (size_t)oc * 4 * 9;
    float bval = bias[oc];
    float mx = -3.4e38f;
    for (int dy = 0; dy < 2; ++dy)
        for (int dx = 0; dx < 2; ++dx) {
            int h = hp * 2 + dy, wx = wp * 2 + dx;
            float s = bval;
            for (int ic = 0; ic < 4; ++ic) {
                int cin = j * CC + LCC + g * 4 + ic;
                for (int kh = 0; kh < 3; ++kh) {
                    int hy = h + kh - 1;
                    if (hy < 0 || hy >= HH) continue;
                    for (int kw = 0; kw < 3; ++kw) {
                        int wy = wx + kw - 1;
                        if (wy < 0 || wy >= WW) continue;
                        s += bf2f(qkv[((size_t)b * NN + hy * WW + wy) * QKVN + cin]) *
                             wk[ic * 9 + kh * 3 + kw];
                    }
                }
            }
            mx = fmaxf(mx, s);
        }
    out[idx] = f2bf(mx);
}

// ---------------- scale-2 grouped conv 3x3 (s2,p2,d2) + maxpool2 ----------------
// channels [512..768); conv out 28x28 -> pooled 14x14; out g2 bf16 (B,3,256,14,14)
__global__ void conv2_pool_kernel(const u16* __restrict__ qkv,
                                  const float* __restrict__ w,
                                  const float* __restrict__ bias,
                                  u16* __restrict__ out)
{
    long idx = (long)blockIdx.x * blockDim.x + threadIdx.x;
    const long total = (long)BB * 3 * LCC * 14 * 14;
    if (idx >= total) return;
    int wp = idx % 14; long t = idx / 14;
    int hp = t % 14;   t /= 14;
    int oc = t % LCC;  t /= LCC;
    int j  = t % 3;    int b = t / 3;
    int g  = oc >> 2;
    const float* wk = w + (size_t)oc * 4 * 9;
    float bval = bias[oc];
    float mx = -3.4e38f;
    for (int dy = 0; dy < 2; ++dy)
        for (int dx = 0; dx < 2; ++dx) {
            int ho = hp * 2 + dy, wo = wp * 2 + dx;   // conv output coords, 0..27
            float s = bval;
            for (int ic = 0; ic < 4; ++ic) {
                int cin = j * CC + 2 * LCC + g * 4 + ic;
                for (int kh = 0; kh < 3; ++kh) {
                    int hy = 2 * ho - 2 + 2 * kh;
                    if (hy < 0 || hy >= HH) continue;
                    for (int kw = 0; kw < 3; ++kw) {
                        int wy = 2 * wo - 2 + 2 * kw;
                        if (wy < 0 || wy >= WW) continue;
                        s += bf2f(qkv[((size_t)b * NN + hy * WW + wy) * QKVN + cin]) *
                             wk[ic * 9 + kh * 3 + kw];
                    }
                }
            }
            mx = fmaxf(mx, s);
        }
    out[idx] = f2bf(mx);
}

// ---------------- windowed attention (one block per (window, head, batch)) ----------------
// mode 0: src = qkv bf16 (B,N,2304), q/k/v at chan offsets 0/768/1536 (+head ch)
// mode 1: src = g bf16 (B,3,256,Hs,Hs)
// writes fused bf16 (B, 3136, 768), token order n = wtile*49 + p (window-major, as ref)
__global__ __launch_bounds__(256) void attn_kernel(
    const u16* __restrict__ src, u16* __restrict__ fused,
    int mode, int Hs, int head_base, int nreg_side, int rep)
{
    const int wr = blockIdx.x;
    const int lh = blockIdx.y;
    const int b  = blockIdx.z;
    const int nreg = nreg_side * nreg_side;

    __shared__ float qS[49 * 64];
    __shared__ float kS[49 * 64];
    __shared__ float vS[49 * 64];
    __shared__ float sS[49 * 49];

    const int tid = threadIdx.x;
    const int wy = wr / nreg_side, wxc = wr % nreg_side;

    for (int idx = tid; idx < 49 * 64; idx += 256) {
        int p = idx >> 6, d = idx & 63;
        int py = p / 7, px = p % 7;
        int hy = wy * 7 + py, wx = wxc * 7 + px;
        float qv, kv, vv;
        if (mode == 0) {
            size_t base = ((size_t)b * NN + hy * WW + wx) * QKVN + lh * HDIM + d;
            qv = bf2f(src[base]);
            kv = bf2f(src[base + CC]);
            vv = bf2f(src[base + 2 * CC]);
        } else {
            size_t cstride = (size_t)Hs * Hs;
            size_t base = ((size_t)b * 3 * LCC + lh * HDIM + d) * cstride + (size_t)hy * Hs + wx;
            qv = bf2f(src[base]);
            kv = bf2f(src[base + (size_t)LCC * cstride]);
            vv = bf2f(src[base + (size_t)2 * LCC * cstride]);
        }
        qS[idx] = qv; kS[idx] = kv; vS[idx] = vv;
    }
    __syncthreads();

    const float scl = 0.125f;   // 64^-0.5
    for (int idx = tid; idx < 49 * 49; idx += 256) {
        int i = idx / 49, jc = idx % 49;
        float s = 0.f;
        #pragma unroll 8
        for (int d = 0; d < 64; ++d) s += qS[i * 64 + d] * kS[jc * 64 + d];
        sS[idx] = s * scl;
    }
    __syncthreads();

    if (tid < 49) {
        float mx = -3.4e38f;
        for (int jc = 0; jc < 49; ++jc) mx = fmaxf(mx, sS[tid * 49 + jc]);
        float sum = 0.f;
        for (int jc = 0; jc < 49; ++jc) {
            float e = __expf(sS[tid * 49 + jc] - mx);
            sS[tid * 49 + jc] = e;
            sum += e;
        }
        float inv = 1.f / sum;
        for (int jc = 0; jc < 49; ++jc) sS[tid * 49 + jc] *= inv;
    }
    __syncthreads();

    for (int idx = tid; idx < 49 * 64; idx += 256) {
        int p = idx >> 6, d = idx & 63;
        float y = 0.f;
        for (int q = 0; q < 49; ++q) y += sS[p * 49 + q] * vS[q * 64 + d];
        u16 yb = f2bf(y);
        for (int r = 0; r < rep; ++r) {
            int wtile = r * nreg + wr;
            size_t n_out = (size_t)wtile * 49 + p;
            fused[((size_t)b * NN + n_out) * CC + (head_base + lh) * HDIM + d] = yb;
        }
    }
}

// ---------------- host orchestration ----------------
extern "C" void kernel_launch(void* const* d_in, const int* in_sizes, int n_in,
                              void* d_out, int out_size, void* d_ws, size_t ws_size,
                              hipStream_t stream)
{
    const float* x       = (const float*)d_in[0];
    const float* w_qkv   = (const float*)d_in[1];
    const float* w_proj  = (const float*)d_in[2];
    const float* b_proj  = (const float*)d_in[3];
    const float* conv1_w = (const float*)d_in[4];
    const float* conv1_b = (const float*)d_in[5];
    const float* conv2_w = (const float*)d_in[6];
    const float* conv2_b = (const float*)d_in[7];
    (void)in_sizes; (void)n_in; (void)out_size; (void)ws_size;

    char* ws = (char*)d_ws;
    size_t off = 0;
    auto walloc = [&](size_t bytes) -> char* {
        char* p = ws + off;
        off += (bytes + 255) & ~(size_t)255;
        return p;
    };

    const long nX    = (long)MTOT * CC;       // 38,535,168
    const long nWqkv = (long)QKVN * CC;       //  1,769,472
    const long nWprj = (long)CC * CC;         //    589,824
    const long nQKV  = (long)MTOT * QKVN;     // 115,605,504
    const long nG1   = (long)BB * 3 * LCC * 28 * 28;
    const long nG2   = (long)BB * 3 * LCC * 14 * 14;

    u16* xbf     = (u16*)walloc((size_t)nX * 2);
    u16* wqkvbf  = (u16*)walloc((size_t)nWqkv * 2);
    u16* wprojbf = (u16*)walloc((size_t)nWprj * 2);
    u16* qkvbf   = (u16*)walloc((size_t)nQKV * 2);
    u16* g1      = (u16*)walloc((size_t)nG1 * 2);
    u16* g2      = (u16*)walloc((size_t)nG2 * 2);
    u16* fused   = xbf;   // reuse: x_bf16 no longer needed after QKV GEMM

    cast_f32_bf16<<<(nX + 255) / 256, 256, 0, stream>>>(x, xbf, nX);
    cast_f32_bf16<<<(nWqkv + 255) / 256, 256, 0, stream>>>(w_qkv, wqkvbf, nWqkv);
    cast_f32_bf16<<<(nWprj + 255) / 256, 256, 0, stream>>>(w_proj, wprojbf, nWprj);

    // QKV GEMM: (50176 x 768) @ (2304 x 768)^T -> bf16 (50176 x 2304)
    wmma_gemm_kernel<false, true><<<dim3(MTOT / BM, QKVN / BN), 256, 0, stream>>>(
        xbf, wqkvbf, nullptr, qkvbf, MTOT, QKVN, CC);

    conv1_pool_kernel<<<(int)((nG1 + 255) / 256), 256, 0, stream>>>(qkvbf, conv1_w, conv1_b, g1);
    conv2_pool_kernel<<<(int)((nG2 + 255) / 256), 256, 0, stream>>>(qkvbf, conv2_w, conv2_b, g2);

    // scale 0: 8x8 windows, heads 0..3, read qkv directly
    attn_kernel<<<dim3(64, 4, BB), 256, 0, stream>>>(qkvbf, fused, 0, 56, 0, 8, 1);
    // scale 1: 4x4 windows on 28x28, heads 4..7, tiled x4
    attn_kernel<<<dim3(16, 4, BB), 256, 0, stream>>>(g1, fused, 1, 28, 4, 4, 4);
    // scale 2: 2x2 windows on 14x14, heads 8..11, tiled x16
    attn_kernel<<<dim3(4, 4, BB), 256, 0, stream>>>(g2, fused, 1, 14, 8, 2, 16);

    // Projection GEMM: (50176 x 768) @ (768 x 768)^T + bias -> f32 out
    wmma_gemm_kernel<true, false><<<dim3(MTOT / BM, CC / BN), 256, 0, stream>>>(
        fused, wprojbf, b_proj, (float*)d_out, MTOT, CC, CC);
}
